// DynamicTransformerLayer_7052336300502
// MI455X (gfx1250) — matrix operand order repaired and checked by reading, not compiled, not run on hardware
//
#include <hip/hip_runtime.h>
#include <hip/hip_bf16.h>
#include <math.h>

#define D_MODEL  1024
#define N_HEADS  16
#define HEAD_DIM 64
#define D_FF     4096
#define RANK     8
#define BATCH    4
#define SEQ      2048
#define MROWS    (BATCH * SEQ)

typedef __attribute__((ext_vector_type(16))) __bf16 v16bf;
typedef __attribute__((ext_vector_type(8)))  float  v8f;
typedef __attribute__((ext_vector_type(4)))  unsigned int u32x4;
typedef __attribute__((ext_vector_type(8)))  int i32x8;
typedef __attribute__((ext_vector_type(4)))  int i32x4;

union FragBF {
    v16bf v;
    uint4 q[2];
    unsigned short u[16];
};

__device__ __forceinline__ unsigned short f2bf(float f) {
    unsigned int u = __float_as_uint(f);
    u += 0x7FFFu + ((u >> 16) & 1u);   // round-to-nearest-even
    return (unsigned short)(u >> 16);
}
__device__ __forceinline__ float bf2f(unsigned short h) {
    return __uint_as_float(((unsigned int)h) << 16);
}

__device__ __forceinline__ float rmax16(float x) {
    x = fmaxf(x, __shfl_xor(x, 1, 32));
    x = fmaxf(x, __shfl_xor(x, 2, 32));
    x = fmaxf(x, __shfl_xor(x, 4, 32));
    x = fmaxf(x, __shfl_xor(x, 8, 32));
    return x;
}
__device__ __forceinline__ float rsum16(float x) {
    x += __shfl_xor(x, 1, 32);
    x += __shfl_xor(x, 2, 32);
    x += __shfl_xor(x, 4, 32);
    x += __shfl_xor(x, 8, 32);
    return x;
}

// ---------------------------------------------------- TDM 2D tile -> LDS
// Loads a (tile_dim0=32 elems) x (tile_dim1=128 rows) bf16 tile, row stride
// K elements, into LDS at `ldsaddr` with 16B padding after each 64B row
// (matches XST=40 ushort row stride).
__device__ __forceinline__ void tdm_load_tile(unsigned ldsaddr,
                                              const void* gaddr, int K,
                                              int Nrows) {
    const unsigned long long ga = (unsigned long long)(uintptr_t)gaddr;
    u32x4 g0;
    g0[0] = 1u;  // count=1, user descriptor
    g0[1] = ldsaddr;
    g0[2] = (unsigned)ga;
    g0[3] = (unsigned)((ga >> 32) & 0x1FFFFFFull) | (2u << 30);  // type=2
    i32x8 g1;
    // data_size=1 (2B) | pad_enable | pad_interval=3 (16 DW) | pad_amount=3 (4 DW)
    g1[0] = (int)((1u << 16) | (1u << 20) | (3u << 22) | (3u << 25));
    g1[1] = (int)(((unsigned)K & 0xFFFFu) << 16);            // tensor_dim0 lo16
    g1[2] = (int)((((unsigned)K >> 16) & 0xFFFFu) |
                  (((unsigned)Nrows & 0xFFFFu) << 16));      // dim0 hi | dim1 lo
    g1[3] = (int)((((unsigned)Nrows >> 16) & 0xFFFFu) |
                  (32u << 16));                              // dim1 hi | tile_dim0
    g1[4] = 128;      // tile_dim1 (tile_dim2 = 0)
    g1[5] = K;        // tensor_dim0_stride lo32
    g1[6] = 0;
    g1[7] = 0;
    const i32x4 z4 = {0, 0, 0, 0};
#if defined(__clang_major__) && (__clang_major__ >= 23)
    const i32x8 z8 = {0, 0, 0, 0, 0, 0, 0, 0};
    __builtin_amdgcn_tensor_load_to_lds(g0, g1, z4, z4, z8, 0);
#else
    __builtin_amdgcn_tensor_load_to_lds(g0, g1, z4, z4, 0);
#endif
}

// ---------------------------------------------------------------- converters
__global__ void f32_to_bf16(const float* __restrict__ in,
                            unsigned short* __restrict__ out, int n) {
    for (int i = blockIdx.x * blockDim.x + threadIdx.x; i < n;
         i += gridDim.x * blockDim.x)
        out[i] = f2bf(in[i]);
}

// ---------------------------------------------------------------- layernorm
__global__ __launch_bounds__(256)
void ln_bf16(const float* __restrict__ x, const float* __restrict__ g,
             const float* __restrict__ beta, unsigned short* __restrict__ out) {
    __shared__ float red[256];
    const int row = blockIdx.x;
    const int tid = threadIdx.x;
    const float4 xv = ((const float4*)(x + (size_t)row * D_MODEL))[tid];
    red[tid] = xv.x + xv.y + xv.z + xv.w;
    __syncthreads();
    for (int st = 128; st > 0; st >>= 1) {
        if (tid < st) red[tid] += red[tid + st];
        __syncthreads();
    }
    const float mu = red[0] * (1.0f / D_MODEL);
    __syncthreads();
    const float d0 = xv.x - mu, d1 = xv.y - mu, d2 = xv.z - mu, d3 = xv.w - mu;
    red[tid] = d0 * d0 + d1 * d1 + d2 * d2 + d3 * d3;
    __syncthreads();
    for (int st = 128; st > 0; st >>= 1) {
        if (tid < st) red[tid] += red[tid + st];
        __syncthreads();
    }
    const float rstd = rsqrtf(red[0] * (1.0f / D_MODEL) + 1e-5f);
    const float4 gv = ((const float4*)g)[tid];
    const float4 bv = ((const float4*)beta)[tid];
    unsigned short* o = out + (size_t)row * D_MODEL + tid * 4;
    o[0] = f2bf(d0 * rstd * gv.x + bv.x);
    o[1] = f2bf(d1 * rstd * gv.y + bv.y);
    o[2] = f2bf(d2 * rstd * gv.z + bv.z);
    o[3] = f2bf(d3 * rstd * gv.w + bv.w);
}

// -------------------------------------------------- rank-8 dyn coefficients
__global__ __launch_bounds__(256)
void coef_kernel(const unsigned short* __restrict__ h,
                 const float* __restrict__ dyn, float* __restrict__ c,
                 int Kdim) {
    __shared__ float red[256];
    const int m = blockIdx.x;
    const int b = m >> 11;  // m / SEQ
    const int tid = threadIdx.x;
    float acc[RANK];
#pragma unroll
    for (int r = 0; r < RANK; r++) acc[r] = 0.0f;
    const unsigned short* hr = h + (size_t)m * Kdim;
    const float* dr = dyn + (size_t)b * RANK * Kdim;
    for (int i = tid; i < Kdim; i += 256) {
        const float hv = bf2f(hr[i]);
#pragma unroll
        for (int r = 0; r < RANK; r++) acc[r] += hv * dr[(size_t)r * Kdim + i];
    }
    for (int r = 0; r < RANK; r++) {
        red[tid] = acc[r];
        __syncthreads();
        for (int st = 128; st > 0; st >>= 1) {
            if (tid < st) red[tid] += red[tid + st];
            __syncthreads();
        }
        if (tid == 0) c[(size_t)m * RANK + r] = red[0];
        __syncthreads();
    }
}

// ---------------------------------------------------------------- WMMA GEMM
// out[m,n] = act( sum_k X[m,k]*W[n,k] + bias[n] + sum_r coef[m,r]*Amat[n,r]
//                 + resid[m,n] )
// MODE 0: fp32 out (+resid).  MODE 1: bf16 scattered to [B,H,T,64].
// MODE 2: exact GELU -> bf16 [M,N].
// X tile: double-buffered GLOBAL_LOAD_ASYNC_TO_LDS_B128 (ASYNCcnt).
// W tile: double-buffered TENSOR_LOAD_TO_LDS (TDM, TENSORcnt), wave 0 issues.
#define BM 128
#define BN 128
#define BK 32
#define XST 40  // LDS row stride (ushorts), keeps 16B alignment (80B rows)

template <int MODE>
__global__ __launch_bounds__(256)
void gemm_wmma(const unsigned short* __restrict__ X,
               const unsigned short* __restrict__ W,
               const float* __restrict__ bias, const float* __restrict__ coef,
               const float* __restrict__ Amat, const float* __restrict__ resid,
               void* __restrict__ outp, int M, int N, int K) {
    __shared__ __align__(16) unsigned short Xs[2][BM][XST];
    __shared__ __align__(16) unsigned short Ws[2][BN][XST];
    __shared__ float Cs[BM][RANK];

    const int tid = threadIdx.x;
    const int lane = tid & 31;
    const int wid = tid >> 5;
    const int wm = wid & 1;   // 2 waves along M
    const int wn = wid >> 1;  // 4 waves along N
    const int bm0 = blockIdx.y * BM;
    const int bn0 = blockIdx.x * BN;
    const int lm = lane & 15;
    const int klo = (lane < 16) ? 0 : 8;
    const int kb16 = (lane < 16) ? 0 : 16;
    const int lrow = tid >> 1;
    const int lhalf = tid & 1;

    if (coef) {
        for (int i = tid; i < BM * RANK; i += 256)
            Cs[i >> 3][i & 7] = coef[(size_t)(bm0 + (i >> 3)) * RANK + (i & 7)];
    }

    // async staging of X: 32B per thread into buffer `buf`
    const unsigned xlds = (unsigned)(uintptr_t)&Xs[0][lrow][lhalf * 16];
    const unsigned wlds0 = (unsigned)(uintptr_t)&Ws[0][0][0];
    const unsigned bufstride = (unsigned)(BM * XST * sizeof(unsigned short));
    const unsigned short* gx0 = X + (size_t)(bm0 + lrow) * K + lhalf * 16;
    const unsigned short* gwtile = W + (size_t)bn0 * K;

#define STAGE_X(buf, k0)                                                       \
    do {                                                                       \
        unsigned xl = xlds + (unsigned)(buf)*bufstride;                        \
        const unsigned short* gx = gx0 + (k0);                                 \
        asm volatile(                                                          \
            "global_load_async_to_lds_b128 %0, %1, off\n\t"                    \
            "global_load_async_to_lds_b128 %0, %1, off offset:16"              \
            :                                                                  \
            : "v"(xl), "v"(gx)                                                 \
            : "memory");                                                       \
    } while (0)

    v8f acc[4][2];
#pragma unroll
    for (int a = 0; a < 4; a++)
#pragma unroll
        for (int b = 0; b < 2; b++)
            acc[a][b] = (v8f){0.f, 0.f, 0.f, 0.f, 0.f, 0.f, 0.f, 0.f};

    const int nk = K / BK;
    STAGE_X(0, 0);
    if (wid == 0) tdm_load_tile(wlds0, gwtile, K, N);

    for (int i = 0; i < nk; i++) {
        asm volatile("s_wait_asynccnt 0x0" ::: "memory");
        if (wid == 0) __builtin_amdgcn_s_wait_tensorcnt(0);
        __syncthreads();
        if (i + 1 < nk) {
            const int nb = (i + 1) & 1;
            STAGE_X(nb, (i + 1) * BK);
            if (wid == 0)
                tdm_load_tile(wlds0 + (unsigned)nb * bufstride,
                              gwtile + (i + 1) * BK, K, N);
        }
        const int buf = i & 1;

        FragBF afr[4], bfr[2];
#pragma unroll
        for (int fm = 0; fm < 4; fm++) {
            const int row = wm * 64 + fm * 16 + lm;
            afr[fm].q[0] = *(const uint4*)&Xs[buf][row][klo];
            afr[fm].q[1] = *(const uint4*)&Xs[buf][row][klo + 16];
        }
#pragma unroll
        for (int fn = 0; fn < 2; fn++) {
            const int row = wn * 32 + fn * 16 + lm;
            bfr[fn].q[0] = *(const uint4*)&Ws[buf][row][klo];
            bfr[fn].q[1] = *(const uint4*)&Ws[buf][row][klo + 16];
        }
#pragma unroll
        for (int fm = 0; fm < 4; fm++)
#pragma unroll
            for (int fn = 0; fn < 2; fn++)
                acc[fm][fn] = __builtin_amdgcn_wmma_f32_16x16x32_bf16(
                    false, afr[fm].v, false, bfr[fn].v, (short)0, acc[fm][fn],
                    false, false);
    }
#undef STAGE_X

    // ---------------- epilogue ----------------
#pragma unroll
    for (int fn = 0; fn < 2; fn++) {
        const int n = bn0 + wn * 32 + fn * 16 + lm;
        const float bs = bias ? bias[n] : 0.0f;
        float Ar[RANK];
        if (Amat) {
            const float4 a0 = *(const float4*)(Amat + (size_t)n * RANK);
            const float4 a1 = *(const float4*)(Amat + (size_t)n * RANK + 4);
            Ar[0] = a0.x; Ar[1] = a0.y; Ar[2] = a0.z; Ar[3] = a0.w;
            Ar[4] = a1.x; Ar[5] = a1.y; Ar[6] = a1.z; Ar[7] = a1.w;
        }
#pragma unroll
        for (int fm = 0; fm < 4; fm++) {
            const int mloc0 = wm * 64 + fm * 16 + ((lane < 16) ? 0 : 8);
#pragma unroll
            for (int v = 0; v < 8; v++) {
                const int mloc = mloc0 + v;
                const int m = bm0 + mloc;
                float val = acc[fm][fn][v] + bs;
                if (coef) {
#pragma unroll
                    for (int r = 0; r < RANK; r++) val += Cs[mloc][r] * Ar[r];
                }
                if (MODE == 0) {
                    if (resid) val += resid[(size_t)m * N + n];
                    ((float*)outp)[(size_t)m * N + n] = val;
                } else if (MODE == 1) {
                    const int b = m >> 11, t = m & (SEQ - 1);
                    const int h = n >> 6, dh = n & 63;
                    ((unsigned short*)outp)[(((size_t)(b * N_HEADS + h) * SEQ + t)
                                             << 6) + dh] = f2bf(val);
                } else {
                    const float ge =
                        0.5f * val * (1.0f + erff(val * 0.70710678118f));
                    ((unsigned short*)outp)[(size_t)m * N + n] = f2bf(ge);
                }
            }
        }
    }
}

// ------------------------------------------------------- flash attention
// Q,K,V: bf16 [B*H, T, 64];  ctx out: bf16 [B*T, 1024] (row-major, heads packed)
__global__ __launch_bounds__(128)
void attn_wmma(const unsigned short* __restrict__ Q,
               const unsigned short* __restrict__ K,
               const unsigned short* __restrict__ V,
               unsigned short* __restrict__ ctx) {
    __shared__ __align__(16) unsigned short Pl[4][16][64];
    __shared__ __align__(16) unsigned short Vt[64][72];

    const int tid = threadIdx.x;
    const int lane = tid & 31;
    const int wv = tid >> 5;
    const int bh = blockIdx.y;
    const int q0 = blockIdx.x * 64;
    const size_t base = (size_t)bh * SEQ * HEAD_DIM;
    const int lm = lane & 15;
    const int klo = (lane < 16) ? 0 : 8;
    const int kb16 = (lane < 16) ? 0 : 16;

    FragBF qf[2];
    {
        const unsigned short* qr =
            Q + base + (size_t)(q0 + wv * 16 + lm) * HEAD_DIM;
        qf[0].q[0] = *(const uint4*)(qr + klo);
        qf[0].q[1] = *(const uint4*)(qr + klo + 16);
        qf[1].q[0] = *(const uint4*)(qr + 32 + klo);
        qf[1].q[1] = *(const uint4*)(qr + 32 + klo + 16);
    }

    v8f ofr[4];
#pragma unroll
    for (int nd = 0; nd < 4; nd++)
        ofr[nd] = (v8f){0.f, 0.f, 0.f, 0.f, 0.f, 0.f, 0.f, 0.f};
    float mrow[8], lrow[8];
#pragma unroll
    for (int v = 0; v < 8; v++) { mrow[v] = -1e30f; lrow[v] = 0.0f; }

    for (int j = 0; j < SEQ; j += 64) {
        __syncthreads();
        for (int i = tid; i < 64 * 64; i += 128) {
            const int key = i >> 6, d = i & 63;
            Vt[d][key] = V[base + (size_t)(j + key) * HEAD_DIM + d];
        }
        __syncthreads();

        // S = Q K^T / 8
        v8f sfr[4];
#pragma unroll
        for (int nf = 0; nf < 4; nf++) {
            FragBF k0f, k1f;
            const unsigned short* kr =
                K + base + (size_t)(j + nf * 16 + lm) * HEAD_DIM;
            k0f.q[0] = *(const uint4*)(kr + klo);
            k0f.q[1] = *(const uint4*)(kr + klo + 16);
            k1f.q[0] = *(const uint4*)(kr + 32 + klo);
            k1f.q[1] = *(const uint4*)(kr + 32 + klo + 16);
            v8f s = (v8f){0.f, 0.f, 0.f, 0.f, 0.f, 0.f, 0.f, 0.f};
            s = __builtin_amdgcn_wmma_f32_16x16x32_bf16(false, qf[0].v, false,
                                                        k0f.v, (short)0, s,
                                                        false, false);
            s = __builtin_amdgcn_wmma_f32_16x16x32_bf16(false, qf[1].v, false,
                                                        k1f.v, (short)0, s,
                                                        false, false);
#pragma unroll
            for (int v = 0; v < 8; v++) s[v] *= 0.125f;
            sfr[nf] = s;
        }

        // online softmax (16-lane row reductions)
#pragma unroll
        for (int v = 0; v < 8; v++) {
            float mx = -1e30f;
#pragma unroll
            for (int nf = 0; nf < 4; nf++) mx = fmaxf(mx, sfr[nf][v]);
            mx = rmax16(mx);
            const float mnew = fmaxf(mrow[v], mx);
            const float corr = __expf(mrow[v] - mnew);
            mrow[v] = mnew;
            float rs = 0.0f;
#pragma unroll
            for (int nf = 0; nf < 4; nf++) {
                const float p = __expf(sfr[nf][v] - mnew);
                sfr[nf][v] = p;
                rs += p;
            }
            rs = rsum16(rs);
            lrow[v] = lrow[v] * corr + rs;
#pragma unroll
            for (int nd = 0; nd < 4; nd++) ofr[nd][v] *= corr;
        }

        // P -> LDS (wave-private), reload in A-fragment layout
#pragma unroll
        for (int nf = 0; nf < 4; nf++)
#pragma unroll
            for (int v = 0; v < 8; v++)
                Pl[wv][v + ((lane < 16) ? 0 : 8)][nf * 16 + lm] =
                    f2bf(sfr[nf][v]);

        FragBF pf[2];
#pragma unroll
        for (int ks = 0; ks < 2; ks++) {
            pf[ks].q[0] = *(const uint4*)&Pl[wv][lm][ks * 32 + klo];
            pf[ks].q[1] = *(const uint4*)&Pl[wv][lm][ks * 32 + klo + 16];
        }
#pragma unroll
        for (int nd = 0; nd < 4; nd++) {
            FragBF v0f, v1f;
            v0f.q[0] = *(const uint4*)&Vt[nd * 16 + lm][klo];
            v0f.q[1] = *(const uint4*)&Vt[nd * 16 + lm][klo + 16];
            v1f.q[0] = *(const uint4*)&Vt[nd * 16 + lm][32 + klo];
            v1f.q[1] = *(const uint4*)&Vt[nd * 16 + lm][32 + klo + 16];
            ofr[nd] = __builtin_amdgcn_wmma_f32_16x16x32_bf16(
                false, pf[0].v, false, v0f.v, (short)0, ofr[nd], false, false);
            ofr[nd] = __builtin_amdgcn_wmma_f32_16x16x32_bf16(
                false, pf[1].v, false, v1f.v, (short)0, ofr[nd], false, false);
        }
    }

    const int b = bh >> 4, h = bh & 15;
#pragma unroll
    for (int nd = 0; nd < 4; nd++) {
        const int d = nd * 16 + lm;
#pragma unroll
        for (int v = 0; v < 8; v++) {
            const int t = q0 + wv * 16 + v + ((lane < 16) ? 0 : 8);
            const float o = ofr[nd][v] / lrow[v];
            ctx[(size_t)(b * SEQ + t) * D_MODEL + h * HEAD_DIM + d] = f2bf(o);
        }
    }
}

// ---------------------------------------------------------------- launcher
extern "C" void kernel_launch(void* const* d_in, const int* in_sizes, int n_in,
                              void* d_out, int out_size, void* d_ws,
                              size_t ws_size, hipStream_t stream) {
    (void)in_sizes; (void)n_in; (void)out_size; (void)ws_size;
    const float* x       = (const float*)d_in[0];
    const float* dyn_q   = (const float*)d_in[1];
    const float* dyn_k   = (const float*)d_in[2];
    const float* dyn_v   = (const float*)d_in[3];
    const float* dyn_ff1 = (const float*)d_in[4];
    const float* dyn_ff2 = (const float*)d_in[5];
    const float* Wq = (const float*)d_in[6];
    const float* Aq = (const float*)d_in[7];
    const float* Wk = (const float*)d_in[8];
    const float* Ak = (const float*)d_in[9];
    const float* Wv = (const float*)d_in[10];
    const float* Av = (const float*)d_in[11];
    const float* Wo = (const float*)d_in[12];
    const float* W1 = (const float*)d_in[13];
    const float* b1 = (const float*)d_in[14];
    const float* A1 = (const float*)d_in[15];
    const float* W2 = (const float*)d_in[16];
    const float* b2 = (const float*)d_in[17];
    const float* A2 = (const float*)d_in[18];
    const float* g1 = (const float*)d_in[19];
    const float* beta1 = (const float*)d_in[20];
    const float* g2 = (const float*)d_in[21];
    const float* beta2 = (const float*)d_in[22];

    char* ws = (char*)d_ws;
    size_t off = 0;
    auto alloc = [&](size_t bytes) -> void* {
        void* p = ws + off;
        off = (off + bytes + 255) & ~(size_t)255;
        return p;
    };
    unsigned short* Wq_bf = (unsigned short*)alloc((size_t)D_MODEL * D_MODEL * 2);
    unsigned short* Wk_bf = (unsigned short*)alloc((size_t)D_MODEL * D_MODEL * 2);
    unsigned short* Wv_bf = (unsigned short*)alloc((size_t)D_MODEL * D_MODEL * 2);
    unsigned short* Wo_bf = (unsigned short*)alloc((size_t)D_MODEL * D_MODEL * 2);
    unsigned short* W1_bf = (unsigned short*)alloc((size_t)D_FF * D_MODEL * 2);
    unsigned short* W2_bf = (unsigned short*)alloc((size_t)D_MODEL * D_FF * 2);
    unsigned short* h1 = (unsigned short*)alloc((size_t)MROWS * D_MODEL * 2);
    unsigned short* h2 = (unsigned short*)alloc((size_t)MROWS * D_MODEL * 2);
    unsigned short* qb = (unsigned short*)alloc((size_t)MROWS * D_MODEL * 2);
    unsigned short* kb = (unsigned short*)alloc((size_t)MROWS * D_MODEL * 2);
    unsigned short* vb = (unsigned short*)alloc((size_t)MROWS * D_MODEL * 2);
    unsigned short* ctxb = (unsigned short*)alloc((size_t)MROWS * D_MODEL * 2);
    unsigned short* ffb = (unsigned short*)alloc((size_t)MROWS * D_FF * 2);
    float* x2 = (float*)alloc((size_t)MROWS * D_MODEL * 4);
    float* cq = (float*)alloc((size_t)MROWS * RANK * 4);
    float* ck = (float*)alloc((size_t)MROWS * RANK * 4);
    float* cv = (float*)alloc((size_t)MROWS * RANK * 4);
    float* c1 = (float*)alloc((size_t)MROWS * RANK * 4);
    float* c2 = (float*)alloc((size_t)MROWS * RANK * 4);

    // weight conversion
    f32_to_bf16<<<2048, 256, 0, stream>>>(Wq, Wq_bf, D_MODEL * D_MODEL);
    f32_to_bf16<<<2048, 256, 0, stream>>>(Wk, Wk_bf, D_MODEL * D_MODEL);
    f32_to_bf16<<<2048, 256, 0, stream>>>(Wv, Wv_bf, D_MODEL * D_MODEL);
    f32_to_bf16<<<2048, 256, 0, stream>>>(Wo, Wo_bf, D_MODEL * D_MODEL);
    f32_to_bf16<<<4096, 256, 0, stream>>>(W1, W1_bf, D_FF * D_MODEL);
    f32_to_bf16<<<4096, 256, 0, stream>>>(W2, W2_bf, D_MODEL * D_FF);

    // LN1 + coefficients
    ln_bf16<<<MROWS, 256, 0, stream>>>(x, g1, beta1, h1);
    coef_kernel<<<MROWS, 256, 0, stream>>>(h1, dyn_q, cq, D_MODEL);
    coef_kernel<<<MROWS, 256, 0, stream>>>(h1, dyn_k, ck, D_MODEL);
    coef_kernel<<<MROWS, 256, 0, stream>>>(h1, dyn_v, cv, D_MODEL);

    // QKV projections (scatter to [B,H,T,64] bf16)
    dim3 gq(D_MODEL / BN, MROWS / BM);
    gemm_wmma<1><<<gq, 256, 0, stream>>>(h1, Wq_bf, nullptr, cq, Aq, nullptr,
                                         qb, MROWS, D_MODEL, D_MODEL);
    gemm_wmma<1><<<gq, 256, 0, stream>>>(h1, Wk_bf, nullptr, ck, Ak, nullptr,
                                         kb, MROWS, D_MODEL, D_MODEL);
    gemm_wmma<1><<<gq, 256, 0, stream>>>(h1, Wv_bf, nullptr, cv, Av, nullptr,
                                         vb, MROWS, D_MODEL, D_MODEL);

    // attention
    dim3 ga(SEQ / 64, BATCH * N_HEADS);
    attn_wmma<<<ga, 128, 0, stream>>>(qb, kb, vb, ctxb);

    // output projection + residual
    gemm_wmma<0><<<gq, 256, 0, stream>>>(ctxb, Wo_bf, nullptr, nullptr, nullptr,
                                         x, x2, MROWS, D_MODEL, D_MODEL);

    // LN2 + FF
    ln_bf16<<<MROWS, 256, 0, stream>>>(x2, g2, beta2, h2);
    coef_kernel<<<MROWS, 256, 0, stream>>>(h2, dyn_ff1, c1, D_MODEL);
    dim3 gf1(D_FF / BN, MROWS / BM);
    gemm_wmma<2><<<gf1, 256, 0, stream>>>(h2, W1_bf, b1, c1, A1, nullptr, ffb,
                                          MROWS, D_FF, D_MODEL);
    coef_kernel<<<MROWS, 256, 0, stream>>>(ffb, dyn_ff2, c2, D_FF);
    gemm_wmma<0><<<gq, 256, 0, stream>>>(ffb, W2_bf, b2, c2, A2, x2,
                                         (float*)d_out, MROWS, D_MODEL, D_FF);
}
